// DeformableConv2d_10144712753385
// MI455X (gfx1250) — matrix operand order repaired and checked by reading, not compiled, not run on hardware
//
#include <hip/hip_runtime.h>
#include <hip/hip_bf16.h>

// ---------------------------------------------------------------------------
// Problem constants (reference: B,C,O,H,W,K = 4,64,64,128,128,3)
// ---------------------------------------------------------------------------
#define BN   4
#define CN   64
#define ON   64
#define HN   128
#define WN   128
#define K2N  9
#define KTOT 576            // C * K2 = 64*9, shared K dimension of both GEMMs
#define TPX  64             // pixels per workgroup tile
#define NTHREADS 128        // 4 waves (wave32)

#define WSTRIDE  592        // padded row stride (bf16 elems) for weights/patch
#define PKSTRIDE 592
#define WBYTES   (64 * WSTRIDE * 2)   // 75776 bytes per 64x576 bf16 matrix

typedef __attribute__((ext_vector_type(16))) __bf16   v16bf;
typedef __attribute__((ext_vector_type(8)))  float    v8f;
typedef __attribute__((ext_vector_type(8)))  unsigned int v8u;

// float -> bf16 (round-to-nearest-even), kept as raw ushort
__device__ __forceinline__ unsigned short f2bf(float f) {
    unsigned int u = __builtin_bit_cast(unsigned int, f);
    u += 0x7FFFu + ((u >> 16) & 1u);
    return (unsigned short)(u >> 16);
}

// ---------------------------------------------------------------------------
// CDNA5 async global->LDS copy (16B per lane), ASYNCcnt-tracked.
// GVS addressing: saddr = 64-bit SGPR base, vaddr = 32-bit per-lane offset.
// ---------------------------------------------------------------------------
__device__ __forceinline__ void async_copy_b128(unsigned lds_addr,
                                                const void* gbase,
                                                unsigned goff) {
    asm volatile("global_load_async_to_lds_b128 %0, %1, %2"
                 :: "v"(lds_addr), "v"(goff),
                    "s"((unsigned long long)(size_t)gbase)
                 : "memory");
}
__device__ __forceinline__ void wait_async0() {
    asm volatile("s_wait_asynccnt 0x0" ::: "memory");
}

// ---------------------------------------------------------------------------
// Fragment loads from LDS as 16-byte DS loads (layouts per ISA 7.12.2).
// A (16x32 bf16): lane m = lane&15; VGPR v: k = (v>=4)*16 + half*8 + (v&3)*2
//   -> two contiguous 8-bf16 runs at k0+half*8 and k0+16+half*8.
// B (32x16 bf16): lane n = lane&15; VGPR v: k = (v>=4)*16 + half*8 + (v&3)*2
//   -> two contiguous 8-bf16 runs at k0+half*8 and k0+16+half*8.
// ---------------------------------------------------------------------------
__device__ __forceinline__ v16bf lds_fragA(const unsigned short* arow, int k0, int half) {
    uint4 lo = *(const uint4*)(arow + k0 + half * 8);
    uint4 hi = *(const uint4*)(arow + k0 + 16 + half * 8);
    v8u u = {lo.x, lo.y, lo.z, lo.w, hi.x, hi.y, hi.z, hi.w};
    return __builtin_bit_cast(v16bf, u);
}
__device__ __forceinline__ v16bf lds_fragB(const unsigned short* prow, int k0, int half) {
    uint4 lo = *(const uint4*)(prow + k0 + half * 8);
    uint4 hi = *(const uint4*)(prow + k0 + 16 + half * 8);
    v8u u = {lo.x, lo.y, lo.z, lo.w, hi.x, hi.y, hi.z, hi.w};
    return __builtin_bit_cast(v16bf, u);
}

// 64x64x576 GEMM per workgroup; wave owns one 16-px N-tile, 4 M-tiles.
__device__ __forceinline__ void gemm64(const unsigned short* __restrict__ wlds,
                                       const unsigned short* __restrict__ patch,
                                       int lane, int wave, v8f acc[4]) {
    const int half = lane >> 4;
    const int mrow = lane & 15;
    const unsigned short* prow = patch + (wave * 16 + (lane & 15)) * PKSTRIDE;
    for (int k0 = 0; k0 < KTOT; k0 += 32) {
        v16bf bfrag = lds_fragB(prow, k0, half);
#pragma unroll
        for (int t = 0; t < 4; ++t) {
            v16bf afrag = lds_fragA(wlds + (t * 16 + mrow) * WSTRIDE, k0, half);
            acc[t] = __builtin_amdgcn_wmma_f32_16x16x32_bf16(
                false, afrag, false, bfrag, (short)0, acc[t], false, false);
        }
    }
}

// ---------------------------------------------------------------------------
// Prep kernel: one-time f32->bf16 conversion of both weight matrices into the
// exact padded LDS image (row stride WSTRIDE, pad zeroed) for async staging.
// ---------------------------------------------------------------------------
__global__ __launch_bounds__(256)
void prep_weights(const float* __restrict__ w_fc1, const float* __restrict__ w_fc2,
                  const float* __restrict__ w_off, const float* __restrict__ w_mod,
                  const float* __restrict__ w_reg,
                  unsigned short* __restrict__ wg_gate,
                  unsigned short* __restrict__ wg_reg) {
    int idx = blockIdx.x * 256 + threadIdx.x;
    if (idx >= 64 * WSTRIDE) return;
    int row = idx / WSTRIDE, k = idx - row * WSTRIDE;
    float g = 0.f, r = 0.f;
    if (k < KTOT) {
        if      (row < 18) g = w_fc1[row * KTOT + k];
        else if (row < 36) g = w_fc2[(row - 18) * KTOT + k];
        else if (row < 54) g = w_off[(row - 36) * KTOT + k];
        else if (row < 63) g = w_mod[(row - 54) * KTOT + k];
        r = w_reg[row * KTOT + k];
    }
    wg_gate[idx] = f2bf(g);
    wg_reg[idx]  = f2bf(r);
}

// Stage one 75776-byte bf16 weight block into LDS via async b128 copies.
__device__ __forceinline__ void stage_weights_async(unsigned short* wlds,
                                                    const unsigned short* wg,
                                                    int tid) {
    unsigned lbase = (unsigned)(size_t)(void*)wlds;   // low 32b = LDS byte addr
    for (unsigned ofs = (unsigned)tid * 16u; ofs < (unsigned)WBYTES;
         ofs += NTHREADS * 16u)
        async_copy_b128(lbase + ofs, wg, ofs);
}

// ---------------------------------------------------------------------------
// Kernel 1: fused 4-conv (im2col GEMM, 63 rows) + VAE gating epilogue.
// ---------------------------------------------------------------------------
__global__ __launch_bounds__(NTHREADS)
void gate_kernel(const float* __restrict__ x,     const float* __restrict__ eps,
                 const unsigned short* __restrict__ wg_gate,
                 const float* __restrict__ b_fc1, const float* __restrict__ b_fc2,
                 const float* __restrict__ b_off, const float* __restrict__ b_mod,
                 float* __restrict__ ws_off, float* __restrict__ ws_mod) {
    extern __shared__ unsigned char smem[];
    unsigned short* wlds  = (unsigned short*)smem;                 // 64x576 bf16
    unsigned short* patch = (unsigned short*)(smem + WBYTES);      // 64x576 bf16
    float*          scr   = (float*)(smem + 2 * WBYTES);           // 4*64*16 f32

    const int tid = threadIdx.x;
    const int b = blockIdx.z, h = blockIdx.y, w0 = blockIdx.x * TPX;

    stage_weights_async(wlds, wg_gate, tid);

    // im2col patch, packed 8 bf16 per ds_store_b128
    for (int base = tid * 8; base < TPX * KTOT; base += NTHREADS * 8) {
        int p = base / KTOT, k0 = base - p * KTOT;     // 576 % 8 == 0
        unsigned uu[4];
#pragma unroll
        for (int i = 0; i < 4; ++i) {
            unsigned short lohi[2];
#pragma unroll
            for (int j = 0; j < 2; ++j) {
                int k = k0 + i * 2 + j;
                int c = k / 9, r9 = k - c * 9, ky = r9 / 3, kx = r9 - ky * 3;
                int hy = h + ky - 1, wx = w0 + p + kx - 1;
                float v = 0.f;
                if (hy >= 0 && hy < HN && wx >= 0 && wx < WN)
                    v = x[((b * CN + c) * HN + hy) * WN + wx];
                lohi[j] = f2bf(v);
            }
            uu[i] = (unsigned)lohi[0] | ((unsigned)lohi[1] << 16);
        }
        *(uint4*)(patch + p * PKSTRIDE + k0) = make_uint4(uu[0], uu[1], uu[2], uu[3]);
    }
    wait_async0();
    __syncthreads();

    const int wave = tid >> 5, lane = tid & 31;
    const int half = lane >> 4, n = lane & 15;
    v8f acc[4] = {};
    gemm64(wlds, patch, lane, wave, acc);

    float* sb = scr + wave * (64 * 16);
#pragma unroll
    for (int t = 0; t < 4; ++t)
#pragma unroll
        for (int r = 0; r < 8; ++r)
            sb[(t * 16 + r + 8 * half) * 16 + n] = acc[t][r];
    __syncthreads();

    const int p = n;
    const int wx = w0 + wave * 16 + p;
#pragma unroll
    for (int i = 0; i < 9; ++i) {
        int j = half * 9 + i;                                   // 0..17
        float mu = fmaxf(sb[j * 16 + p] + b_fc1[j], 0.f);
        float lv = fmaxf(sb[(18 + j) * 16 + p] + b_fc2[j], 0.f);
        float of = sb[(36 + j) * 16 + p] + b_off[j];
        float e  = eps[((b * 18 + j) * HN + h) * WN + wx];
        float z  = e * __expf(0.5f * lv) + mu;
        float gate = 1.f / (1.f + __expf(-z));
        ws_off[((b * 18 + j) * HN + h) * WN + wx] = of * gate;
        if (half == 0) {
            float mr = sb[(54 + i) * 16 + p] + b_mod[i];
            ws_mod[((b * K2N + i) * HN + h) * WN + wx] = 2.f / (1.f + __expf(-mr));
        }
    }
}

// ---------------------------------------------------------------------------
// Kernel 2: deformable bilinear sampling + 64x64x576 WMMA GEMM with w_reg.
// ---------------------------------------------------------------------------
struct SampP { float w00, w01, w10, w11; short y0, y1, x0, x1; };

__global__ __launch_bounds__(NTHREADS)
void deform_kernel(const float* __restrict__ x,
                   const unsigned short* __restrict__ wg_reg,
                   const float* __restrict__ ws_off, const float* __restrict__ ws_mod,
                   float* __restrict__ out) {
    extern __shared__ unsigned char smem[];
    unsigned short* wlds  = (unsigned short*)smem;
    unsigned short* patch = (unsigned short*)(smem + WBYTES);
    SampP*          prm   = (SampP*)(smem + 2 * WBYTES);           // 576 entries

    const int tid = threadIdx.x;
    const int b = blockIdx.z, h = blockIdx.y, w0 = blockIdx.x * TPX;

    stage_weights_async(wlds, wg_reg, tid);

    // Per (pixel, tap): bilinear corner weights with mask + validity folded in
    for (int idx = tid; idx < TPX * K2N; idx += NTHREADS) {
        int p = idx / K2N, j = idx - p * K2N;
        int wx = w0 + p;
        float oy = ws_off[((b * 18 + 2 * j) * HN + h) * WN + wx];
        float ox = ws_off[((b * 18 + 2 * j + 1) * HN + h) * WN + wx];
        float m  = ws_mod[((b * K2N + j) * HN + h) * WN + wx];
        float py = (float)(h + j / 3 - 1) + oy;
        float px = (float)(wx + (j % 3) - 1) + ox;
        float y0f = floorf(py), x0f = floorf(px);
        float ly = py - y0f, lx = px - x0f;
        int y0 = (int)y0f, xx0 = (int)x0f;
        int y1 = y0 + 1,  xx1 = xx0 + 1;
        float vy0 = (y0 >= 0 && y0 < HN) ? 1.f : 0.f;
        float vy1 = (y1 >= 0 && y1 < HN) ? 1.f : 0.f;
        float vx0 = (xx0 >= 0 && xx0 < WN) ? 1.f : 0.f;
        float vx1 = (xx1 >= 0 && xx1 < WN) ? 1.f : 0.f;
        SampP s;
        s.w00 = (1.f - ly) * (1.f - lx) * vy0 * vx0 * m;
        s.w01 = (1.f - ly) * lx         * vy0 * vx1 * m;
        s.w10 = ly * (1.f - lx)         * vy1 * vx0 * m;
        s.w11 = ly * lx                 * vy1 * vx1 * m;
        s.y0 = (short)min(max(y0, 0), HN - 1);
        s.y1 = (short)min(max(y1, 0), HN - 1);
        s.x0 = (short)min(max(xx0, 0), WN - 1);
        s.x1 = (short)min(max(xx1, 0), WN - 1);
        prm[idx] = s;
    }
    __syncthreads();

    // Sampled patch: patch[p][c*9+j] = bilinear(x[b,c], tap j of pixel p),
    // packed 8 bf16 per ds_store_b128.
    for (int base = tid * 8; base < TPX * KTOT; base += NTHREADS * 8) {
        int p = base / KTOT, k0 = base - p * KTOT;
        unsigned uu[4];
#pragma unroll
        for (int i = 0; i < 4; ++i) {
            unsigned short lohi[2];
#pragma unroll
            for (int jj = 0; jj < 2; ++jj) {
                int k = k0 + i * 2 + jj;
                int c = k / K2N, j = k - c * K2N;
                SampP s = prm[p * K2N + j];
                const float* pl = x + (size_t)(b * CN + c) * (HN * WN);
                float v = s.w00 * pl[(int)s.y0 * WN + (int)s.x0]
                        + s.w01 * pl[(int)s.y0 * WN + (int)s.x1]
                        + s.w10 * pl[(int)s.y1 * WN + (int)s.x0]
                        + s.w11 * pl[(int)s.y1 * WN + (int)s.x1];
                lohi[jj] = f2bf(v);
            }
            uu[i] = (unsigned)lohi[0] | ((unsigned)lohi[1] << 16);
        }
        *(uint4*)(patch + p * PKSTRIDE + k0) = make_uint4(uu[0], uu[1], uu[2], uu[3]);
    }
    wait_async0();
    __syncthreads();

    const int wave = tid >> 5, lane = tid & 31;
    const int half = lane >> 4, n = lane & 15;
    v8f acc[4] = {};
    gemm64(wlds, patch, lane, wave, acc);

    const int wx = w0 + wave * 16 + n;
#pragma unroll
    for (int t = 0; t < 4; ++t)
#pragma unroll
        for (int r = 0; r < 8; ++r) {
            int o = t * 16 + r + 8 * half;
            out[((b * ON + o) * HN + h) * WN + wx] = acc[t][r];
        }
}

// ---------------------------------------------------------------------------
// Host launcher
// ---------------------------------------------------------------------------
extern "C" void kernel_launch(void* const* d_in, const int* in_sizes, int n_in,
                              void* d_out, int out_size, void* d_ws, size_t ws_size,
                              hipStream_t stream) {
    const float* x     = (const float*)d_in[0];
    const float* eps   = (const float*)d_in[1];
    const float* w_fc1 = (const float*)d_in[2];
    const float* b_fc1 = (const float*)d_in[3];
    const float* w_fc2 = (const float*)d_in[4];
    const float* b_fc2 = (const float*)d_in[5];
    const float* w_off = (const float*)d_in[6];
    const float* b_off = (const float*)d_in[7];
    const float* w_mod = (const float*)d_in[8];
    const float* b_mod = (const float*)d_in[9];
    const float* w_reg = (const float*)d_in[10];
    float* out = (float*)d_out;

    // Workspace layout (16B-aligned blocks):
    unsigned short* wg_gate = (unsigned short*)d_ws;               // 75776 B
    unsigned short* wg_reg  = wg_gate + 64 * WSTRIDE;              // 75776 B
    float* ws_off = (float*)(wg_reg + 64 * WSTRIDE);               // B*18*H*W f32
    float* ws_mod = ws_off + (size_t)BN * 18 * HN * WN;            // B*9*H*W f32

    int prep_elems = 64 * WSTRIDE;
    prep_weights<<<(prep_elems + 255) / 256, 256, 0, stream>>>(
        w_fc1, w_fc2, w_off, w_mod, w_reg, wg_gate, wg_reg);

    dim3 grid(WN / TPX, HN, BN);                                   // 2 x 128 x 4
    size_t lds1 = 2 * WBYTES + 4 * 64 * 16 * sizeof(float);        // 167936 B
    size_t lds2 = 2 * WBYTES + TPX * K2N * sizeof(SampP);          // 165376 B

    gate_kernel<<<grid, NTHREADS, lds1, stream>>>(
        x, eps, wg_gate, b_fc1, b_fc2, b_off, b_mod, ws_off, ws_mod);
    deform_kernel<<<grid, NTHREADS, lds2, stream>>>(
        x, wg_reg, ws_off, ws_mod, out);
}
